// SpaceTempGoG_detr_dad_4973572128883
// MI455X (gfx1250) — compile-verified, matrix operand order checked by reading, not run on hardware
//
#include <hip/hip_runtime.h>

typedef __attribute__((ext_vector_type(16))) __bf16 v16bf;
typedef __attribute__((ext_vector_type(8)))  __bf16 v8bf;
typedef __attribute__((ext_vector_type(8)))  float  v8f;

// ---------------- problem constants (match reference) ----------------
static constexpr int NN   = 30000;   // nodes
static constexpr int NPG  = 100;     // nodes per graph
static constexpr int BGR  = 300;     // graphs (= T frames)
static constexpr int ESN  = 500000;  // spatial edges
static constexpr int ETN  = 500000;  // temporal edges
static constexpr int DIN  = 2048;
static constexpr int DLBL = 300;
static constexpr int DE   = 128;
static constexpr int D2   = 256;
static constexpr int DH   = 64;
static constexpr int RK   = 80;

// ---------------- WMMA GEMM: C[M,N] = A[M,K](f32) @ B[K,N](f32) + bias ----------------
static constexpr int BM = 128, BN = 128, BK = 32, PADB = 8, BKP = BK + PADB;

__device__ inline v16bf frag_ld(const __bf16* __restrict__ base) {
  const int lane = threadIdx.x & 31;
  const __bf16* p = base + (size_t)(lane & 15) * BKP;
  const int kg = (lane >> 4) << 3;
  v16bf f;
#pragma unroll
  for (int v = 0; v < 8; ++v) {
    const int k = ((v >> 2) << 4) + kg + ((v & 3) << 1);
    f[2 * v]     = p[k];
    f[2 * v + 1] = p[k + 1];
  }
  return f;
}

// 16 f32 -> 16 bf16, written as two 16-byte LDS stores (dst 16B-aligned)
__device__ inline void stage_store(const float* __restrict__ v, __bf16* dst) {
  v8bf p0, p1;
#pragma unroll
  for (int j = 0; j < 8; ++j) { p0[j] = (__bf16)v[j]; p1[j] = (__bf16)v[8 + j]; }
  *(v8bf*)dst       = p0;
  *(v8bf*)(dst + 8) = p1;
}

// transB==0: B is [K,N] row-major (ldb). transB==1: B is [N,K] row-major (ldb).
// act: 0 none, 1 relu.
__global__ void __launch_bounds__(256)
gemm_wmma_bf16(const float* __restrict__ A, int lda, int colA,
               const float* __restrict__ B, int ldb, int transB,
               const float* __restrict__ bias,
               float* __restrict__ C, int ldc,
               int M, int N, int K, int act) {
  __shared__ __bf16 As[BM][BKP];
  __shared__ __bf16 Bs[BN][BKP];

  const int tid  = threadIdx.x;
  const int lane = tid & 31;
  const int wid  = tid >> 5;
  const int wm   = wid >> 1;          // 0..3  -> 32 rows each
  const int wn   = wid & 1;           // 0..1  -> 64 cols each
  const int m0   = blockIdx.y * BM;
  const int n0   = blockIdx.x * BN;

  v8f acc[2][4] = {};

  const int lr = tid >> 1;            // 0..127 (row of A tile / row of Bs)
  const int lc = (tid & 1) * 16;      // 0 or 16

  // OOB rows of A / cols of B feed only C rows/cols that are never stored,
  // so clamp the index (stay in the allocation) and load UNconditionally.
  const int ra = min(m0 + lr, M - 1);
  const int rb = min(n0 + lr, N - 1);
  const float* __restrict__ Ap  = A + (size_t)ra * lda + colA + lc;  // k index = lc + i
  const float* __restrict__ Bpt = B + (size_t)rb * ldb + lc;         // transB: k index = lc + i
  const float* __restrict__ Bpn = B + rb;                            // !transB: stride ldb per k

  const int kfull = K & ~(BK - 1);
  float va[16], vb[16];

  for (int k0 = 0; k0 < kfull; k0 += BK) {
    // ---- stage tiles: straight-line, no predication ----
#pragma unroll
    for (int j = 0; j < 16; ++j) va[j] = Ap[k0 + j];
    if (transB) {
#pragma unroll
      for (int j = 0; j < 16; ++j) vb[j] = Bpt[k0 + j];
    } else {
#pragma unroll
      for (int j = 0; j < 16; ++j) vb[j] = Bpn[(size_t)(k0 + lc + j) * ldb];
    }
    if (k0 + BK < kfull) {
      __builtin_prefetch(&Ap[k0 + BK], 0, 1);
      if (transB) __builtin_prefetch(&Bpt[k0 + BK], 0, 1);
      else        __builtin_prefetch(&Bpn[(size_t)(k0 + BK + lc) * ldb], 0, 1);
    }
    stage_store(va, &As[lr][lc]);
    stage_store(vb, &Bs[lr][lc]);
    __syncthreads();

    // ---- WMMA on the tile ----
    v16bf af[2], bf[4];
#pragma unroll
    for (int mi = 0; mi < 2; ++mi) af[mi] = frag_ld(&As[wm * 32 + mi * 16][0]);
#pragma unroll
    for (int ni = 0; ni < 4; ++ni) bf[ni] = frag_ld(&Bs[wn * 64 + ni * 16][0]);
#pragma unroll
    for (int mi = 0; mi < 2; ++mi)
#pragma unroll
      for (int ni = 0; ni < 4; ++ni)
        acc[mi][ni] = __builtin_amdgcn_wmma_f32_16x16x32_bf16(
            false, af[mi], false, bf[ni], (short)0, acc[mi][ni], false, false);
    __syncthreads();
  }

  // ---- K remainder tile (needs zero fill along K); runs at most once ----
  if (kfull < K) {
#pragma unroll
    for (int j = 0; j < 16; ++j) {
      const int kk = kfull + lc + j;
      const float a = Ap[max(min(kfull + j, K - 1 - lc), 0)];
      va[j] = (kk < K) ? a : 0.0f;
    }
    if (transB) {
#pragma unroll
      for (int j = 0; j < 16; ++j) {
        const int kk = kfull + lc + j;
        const float b = Bpt[max(min(kfull + j, K - 1 - lc), 0)];
        vb[j] = (kk < K) ? b : 0.0f;
      }
    } else {
#pragma unroll
      for (int j = 0; j < 16; ++j) {
        const int kk = kfull + lc + j;
        const float b = Bpn[(size_t)min(kk, K - 1) * ldb];
        vb[j] = (kk < K) ? b : 0.0f;
      }
    }
    stage_store(va, &As[lr][lc]);
    stage_store(vb, &Bs[lr][lc]);
    __syncthreads();

    v16bf af[2], bf[4];
#pragma unroll
    for (int mi = 0; mi < 2; ++mi) af[mi] = frag_ld(&As[wm * 32 + mi * 16][0]);
#pragma unroll
    for (int ni = 0; ni < 4; ++ni) bf[ni] = frag_ld(&Bs[wn * 64 + ni * 16][0]);
#pragma unroll
    for (int mi = 0; mi < 2; ++mi)
#pragma unroll
      for (int ni = 0; ni < 4; ++ni)
        acc[mi][ni] = __builtin_amdgcn_wmma_f32_16x16x32_bf16(
            false, af[mi], false, bf[ni], (short)0, acc[mi][ni], false, false);
  }

  // ---- store C (VGPR e -> m = e + 8*(lane>=16), n = lane&15) ----
  const int rbs = (lane >> 4) * 8;
  const int cc  = lane & 15;
  if (m0 + BM <= M && n0 + BN <= N) {
    // interior block: no guards
#pragma unroll
    for (int mi = 0; mi < 2; ++mi)
#pragma unroll
      for (int ni = 0; ni < 4; ++ni) {
        const int gc = n0 + wn * 64 + ni * 16 + cc;
        const float bv = bias ? bias[gc] : 0.0f;
#pragma unroll
        for (int e = 0; e < 8; ++e) {
          const int gr = m0 + wm * 32 + mi * 16 + rbs + e;
          float o = acc[mi][ni][e] + bv;
          if (act == 1) o = fmaxf(o, 0.0f);
          C[(size_t)gr * ldc + gc] = o;
        }
      }
  } else {
#pragma unroll
    for (int mi = 0; mi < 2; ++mi)
#pragma unroll
      for (int ni = 0; ni < 4; ++ni) {
        const int gc = n0 + wn * 64 + ni * 16 + cc;
        if (gc >= N) continue;
        const float bv = bias ? bias[gc] : 0.0f;
#pragma unroll
        for (int e = 0; e < 8; ++e) {
          const int gr = m0 + wm * 32 + mi * 16 + rbs + e;
          if (gr < M) {
            float o = acc[mi][ni][e] + bv;
            if (act == 1) o = fmaxf(o, 0.0f);
            C[(size_t)gr * ldc + gc] = o;
          }
        }
      }
  }
}

// ---------------- small helper kernels ----------------
__global__ void fill_f32(float* p, float v, int n) {
  int i = blockIdx.x * 256 + threadIdx.x;
  if (i < n) p[i] = v;
}

// per-column mean / inv-std over M rows (biased var), grid.x = D, block 256
__global__ void colstats(const float* __restrict__ X, int ld, int M,
                         float* mean, float* rstd) {
  const int col = blockIdx.x;
  float s = 0.f, s2 = 0.f;
  for (int r = threadIdx.x; r < M; r += 256) {
    float v = X[(size_t)r * ld + col];
    s += v; s2 += v * v;
  }
  __shared__ float sh[256], sh2[256];
  sh[threadIdx.x] = s; sh2[threadIdx.x] = s2; __syncthreads();
  for (int o = 128; o > 0; o >>= 1) {
    if (threadIdx.x < o) { sh[threadIdx.x] += sh[threadIdx.x + o]; sh2[threadIdx.x] += sh2[threadIdx.x + o]; }
    __syncthreads();
  }
  if (threadIdx.x == 0) {
    float m = sh[0] / M;
    float var = sh2[0] / M - m * m;
    mean[col] = m;
    rstd[col] = rsqrtf(var + 1e-5f);
  }
}

// out[r, off+c] = lrelu(norm(X[r,c])); gamma==null -> no affine (instance norm)
__global__ void norm_apply(const float* __restrict__ X, int D, int M,
                           const float* mean, const float* rstd,
                           const float* gamma, const float* beta,
                           float* __restrict__ out, int ldo, int off) {
  int i = blockIdx.x * 256 + threadIdx.x;
  if (i >= M * D) return;
  int r = i / D, c = i % D;
  float v = (X[i] - mean[c]) * rstd[c];
  if (gamma) v = v * gamma[c] + beta[c];
  v = v >= 0.f ? v : 0.2f * v;
  out[(size_t)r * ldo + off + c] = v;
}

// monotone float<->uint key for atomicMax-based segment max
__device__ inline unsigned fkey(float x) {
  unsigned b = __float_as_uint(x);
  return (b & 0x80000000u) ? ~b : (b | 0x80000000u);
}
__device__ inline float funkey(unsigned k) {
  return (k & 0x80000000u) ? __uint_as_float(k ^ 0x80000000u) : __uint_as_float(~k);
}

// logit_e = dot(q[dst], k[src] + w_e*We)/8 ; segment max via keyed atomicMax
__global__ void edge_logit(const int* __restrict__ src, const int* __restrict__ dst,
                           const float* __restrict__ q, const float* __restrict__ k,
                           const float* __restrict__ ew, int ews,
                           const float* __restrict__ We,
                           float* __restrict__ logit, unsigned* __restrict__ nmax, int E) {
  int e = blockIdx.x * 8 + (threadIdx.x >> 5);
  if (e >= E) return;
  int lane = threadIdx.x & 31;
  int s = src[e], d = dst[e];
  float w = ew[(size_t)e * ews];
  float p = 0.f;
  for (int j = lane; j < DH; j += 32)
    p += q[(size_t)d * DH + j] * (k[(size_t)s * DH + j] + w * We[j]);
  for (int o = 16; o; o >>= 1) p += __shfl_down(p, o, 32);
  if (lane == 0) {
    float lg = p * 0.125f;  // 1/sqrt(64)
    logit[e] = lg;
    atomicMax(&nmax[d], fkey(lg));
  }
}

__global__ void edge_expsum(const int* __restrict__ dst, const float* __restrict__ logit,
                            const unsigned* __restrict__ nmax,
                            float* __restrict__ w, float* __restrict__ denom, int E) {
  int e = blockIdx.x * 256 + threadIdx.x;
  if (e >= E) return;
  int d = dst[e];
  float x = __expf(logit[e] - funkey(nmax[d]));
  w[e] = x;
  atomicAdd(&denom[d], x);
}

__global__ void edge_scatter(const int* __restrict__ src, const int* __restrict__ dst,
                             const float* __restrict__ v,
                             const float* __restrict__ ew, int ews,
                             const float* __restrict__ We,
                             const float* __restrict__ w, const float* __restrict__ denom,
                             float* __restrict__ out, int E) {
  int e = blockIdx.x * 8 + (threadIdx.x >> 5);
  if (e >= E) return;
  int lane = threadIdx.x & 31;
  int s = src[e], d = dst[e];
  float alpha = w[e] / denom[d];
  float wt = ew[(size_t)e * ews];
  for (int j = lane; j < DH; j += 32)
    atomicAdd(&out[(size_t)d * DH + j], alpha * (v[(size_t)s * DH + j] + wt * We[j]));
}

// beta = sigmoid([out, xr, out-xr] @ Wb) ; res = beta*xr + (1-beta)*out  (in-place ok)
__global__ void beta_gate(const float* __restrict__ out, const float* __restrict__ xr,
                          const float* __restrict__ Wb, float* __restrict__ res, int Nn) {
  int i = blockIdx.x * 8 + (threadIdx.x >> 5);
  if (i >= Nn) return;
  int lane = threadIdx.x & 31;
  float p = 0.f;
  for (int j = lane; j < DH; j += 32) {
    float o = out[(size_t)i * DH + j], x = xr[(size_t)i * DH + j];
    p += o * Wb[j] + x * Wb[DH + j] + (o - x) * Wb[2 * DH + j];
  }
  for (int o = 16; o; o >>= 1) p += __shfl_down(p, o, 32);
  float beta = 1.f / (1.f + __expf(-__shfl(p, 0, 32)));
  for (int j = lane; j < DH; j += 32) {
    float o = out[(size_t)i * DH + j], x = xr[(size_t)i * DH + j];
    res[(size_t)i * DH + j] = beta * x + (1.f - beta) * o;
  }
}

__global__ void pool_agg(const int* __restrict__ src, const int* __restrict__ dst,
                         const float* __restrict__ n, float* __restrict__ agg, int E) {
  int e = blockIdx.x * 8 + (threadIdx.x >> 5);
  if (e >= E) return;
  int lane = threadIdx.x & 31;
  int s = src[e], d = dst[e];
  for (int j = lane; j < DE; j += 32)
    atomicAdd(&agg[(size_t)d * DE + j], n[(size_t)s * DE + j]);
}

__global__ void pool_score(const float* __restrict__ agg, const float* __restrict__ n,
                           const float* __restrict__ Wrel, const float* __restrict__ brel,
                           const float* __restrict__ Wroot, float* __restrict__ score, int Nn) {
  int i = blockIdx.x * 8 + (threadIdx.x >> 5);
  if (i >= Nn) return;
  int lane = threadIdx.x & 31;
  float p = 0.f;
  for (int j = lane; j < DE; j += 32)
    p += agg[(size_t)i * DE + j] * Wrel[j] + n[(size_t)i * DE + j] * Wroot[j];
  for (int o = 16; o; o >>= 1) p += __shfl_down(p, o, 32);
  if (lane == 0) score[i] = tanhf(p + brel[0]);
}

// kth[b] = 80th largest of the 100 scores in graph b
__global__ void pool_kth(const float* __restrict__ score, float* __restrict__ kth) {
  __shared__ float s[NPG];
  __shared__ float red[128];
  int b = blockIdx.x, t = threadIdx.x;
  if (t < NPG) s[t] = score[b * NPG + t];
  __syncthreads();
  float cand = 3.4028235e38f;
  if (t < NPG) {
    int gt = 0; float mine = s[t];
    for (int j = 0; j < NPG; ++j) gt += (s[j] > mine);
    if (gt < RK) cand = mine;
  }
  red[t] = cand; __syncthreads();
  for (int o = 64; o; o >>= 1) { if (t < o) red[t] = fminf(red[t], red[t + o]); __syncthreads(); }
  if (t == 0) kth[b] = red[0];
}

// g[b,c] = max over kept nodes of n[i,c]*s[i]
__global__ void pool_readout(const float* __restrict__ n, const float* __restrict__ score,
                             const float* __restrict__ kth, float* __restrict__ g) {
  int b = blockIdx.x, c = threadIdx.x;  // 128 threads
  float kv = kth[b];
  float m = -3.4028235e38f;
  for (int j = 0; j < NPG; ++j) {
    float sc = score[b * NPG + j];
    if (sc >= kv) m = fmaxf(m, n[(size_t)(b * NPG + j) * DE + c] * sc);
  }
  g[b * DE + c] = m;
}

// row softmax with scale, grid.x = rows, block 256
__global__ void softmax_rows(float* __restrict__ X, int Nc, float scale) {
  int r = blockIdx.x, t = threadIdx.x;
  __shared__ float sh[256];
  float m = -3.4028235e38f;
  for (int j = t; j < Nc; j += 256) m = fmaxf(m, X[(size_t)r * Nc + j] * scale);
  sh[t] = m; __syncthreads();
  for (int o = 128; o; o >>= 1) { if (t < o) sh[t] = fmaxf(sh[t], sh[t + o]); __syncthreads(); }
  m = sh[0]; __syncthreads();
  float s = 0.f;
  for (int j = t; j < Nc; j += 256) {
    float e = __expf(X[(size_t)r * Nc + j] * scale - m);
    X[(size_t)r * Nc + j] = e; s += e;
  }
  sh[t] = s; __syncthreads();
  for (int o = 128; o; o >>= 1) { if (t < o) sh[t] += sh[t + o]; __syncthreads(); }
  s = sh[0];
  for (int j = t; j < Nc; j += 256) X[(size_t)r * Nc + j] /= s;
}

// out = LayerNorm(x + r) row-wise over D2, block == D2 == 256
__global__ void resid_ln(const float* __restrict__ x, const float* __restrict__ r,
                         const float* __restrict__ g, const float* __restrict__ b,
                         float* __restrict__ out) {
  int row = blockIdx.x, t = threadIdx.x;
  float v = x[(size_t)row * D2 + t] + r[(size_t)row * D2 + t];
  __shared__ float sh[256];
  sh[t] = v; __syncthreads();
  for (int o = 128; o; o >>= 1) { if (t < o) sh[t] += sh[t + o]; __syncthreads(); }
  float mean = sh[0] / D2; __syncthreads();
  float d = v - mean;
  sh[t] = d * d; __syncthreads();
  for (int o = 128; o; o >>= 1) { if (t < o) sh[t] += sh[t + o]; __syncthreads(); }
  float var = sh[0] / D2;
  out[(size_t)row * D2 + t] = d * rsqrtf(var + 1e-5f) * g[t] + b[t];
}

// persistent-block LSTM: X[T, 4D] precomputed x@Wih, recur h@Whh in-kernel.
// blockDim.x == 4*D ; torch gate order i,f,g,o.
__global__ void __launch_bounds__(1024)
lstm_seq(const float* __restrict__ X, const float* __restrict__ Whh,
         const float* __restrict__ bih, const float* __restrict__ bhh,
         int T, int D, float* __restrict__ hout) {
  __shared__ float z[1024], h[256], c[256];
  const int t = threadIdx.x;
  const int G = blockDim.x;  // 4*D
  if (t < D) { h[t] = 0.f; c[t] = 0.f; }
  __syncthreads();
  const float bsum = bih[t] + bhh[t];
  for (int step = 0; step < T; ++step) {
    float s = X[(size_t)step * G + t] + bsum;
    for (int j = 0; j < D; ++j) s += h[j] * Whh[(size_t)j * G + t];
    z[t] = s; __syncthreads();
    if (t < D) {
      float ig = 1.f / (1.f + __expf(-z[t]));
      float fg = 1.f / (1.f + __expf(-z[D + t]));
      float gg = tanhf(z[2 * D + t]);
      float og = 1.f / (1.f + __expf(-z[3 * D + t]));
      float cn = fg * c[t] + ig * gg;
      c[t] = cn;
      h[t] = og * tanhf(cn);
    }
    __syncthreads();
  }
  if (t < D) hout[t] = h[t];
}

// fc1 (640->128, lrelu), fc2 (128->2), softmax. block == 128.
__global__ void final_fc(const float* __restrict__ hg, const float* __restrict__ hi,
                         const float* __restrict__ ha,
                         const float* __restrict__ W1, const float* __restrict__ b1,
                         const float* __restrict__ W2, const float* __restrict__ b2,
                         float* __restrict__ out) {
  __shared__ float cat[640];
  __shared__ float f[128];
  int t = threadIdx.x;
  cat[t]       = hg[t];
  cat[128 + t] = hi[t];
  cat[256 + t] = hi[128 + t];
  cat[384 + t] = ha[t];
  cat[512 + t] = ha[128 + t];
  __syncthreads();
  float s = b1[t];
  for (int j = 0; j < 640; ++j) s += cat[j] * W1[(size_t)j * 128 + t];
  f[t] = s >= 0.f ? s : 0.2f * s;
  __syncthreads();
  if (t == 0) {
    float l0 = b2[0], l1 = b2[1];
    for (int j = 0; j < 128; ++j) { l0 += f[j] * W2[j * 2]; l1 += f[j] * W2[j * 2 + 1]; }
    float m = fmaxf(l0, l1);
    float e0 = __expf(l0 - m), e1 = __expf(l1 - m);
    float inv = 1.f / (e0 + e1);
    out[0] = l0; out[1] = l1; out[2] = e0 * inv; out[3] = e1 * inv;
  }
}

// ---------------- host-side orchestration ----------------
static inline int ceil_div(int a, int b) { return (a + b - 1) / b; }

static void gemm(hipStream_t st, const float* A, int lda, int colA,
                 const float* B, int ldb, int transB, const float* bias,
                 float* C, int ldc, int M, int N, int K, int act) {
  dim3 g(ceil_div(N, BN), ceil_div(M, BM));
  gemm_wmma_bf16<<<g, dim3(256), 0, st>>>(A, lda, colA, B, ldb, transB, bias, C, ldc, M, N, K, act);
}

static void fill(hipStream_t st, float* p, float v, int n) {
  fill_f32<<<ceil_div(n, 256), 256, 0, st>>>(p, v, n);
}

extern "C" void kernel_launch(void* const* d_in, const int* in_sizes, int n_in,
                              void* d_out, int out_size, void* d_ws, size_t ws_size,
                              hipStream_t stream) {
  (void)in_sizes; (void)n_in; (void)out_size; (void)ws_size;

  const float* x    = (const float*)d_in[0];
  const int*   ei   = (const int*)d_in[1];     // [2, ES]
  const float* imgf = (const float*)d_in[2];
  const float* attf = (const float*)d_in[3];
  const float* eemb = (const float*)d_in[4];   // [ES, 4]
  const int*   tadj = (const int*)d_in[5];     // [2, ET]
  const float* tw   = (const float*)d_in[6];
  auto PP = [&](int i) { return (const float*)d_in[8 + i]; };
  // param indices (insertion order of params dict)
  // 0 x_fc_W 1 x_fc_b 2 x_bn_g 3 x_bn_b 4 l_fc_W 5 l_fc_b 6 l_bn_g 7 l_bn_b
  // sp: 8 Wq 9 bq 10 Wk 11 bk 12 Wv 13 bv 14 We 15 Ws 16 bs 17 Wb ; tp: 18..27
  // 28 pool_Wrel 29 pool_brel 30 pool_Wroot
  // 31 img_fc_W 32 img_fc_b 33 att_fc_W 34 att_fc_b
  // timg: 35 inW 36 inb 37 outW 38 outb 39 W1 40 b1 41 W2 42 b2 43 ln1g 44 ln1b 45 ln2g 46 ln2b ; tatt: 47..58
  // lg: 59..62 li: 63..66 la: 67..70 ; 71 fc1_W 72 fc1_b 73 fc2_W 74 fc2_b

  // workspace bump allocation (floats)
  float* Wsp = (float*)d_ws;
  size_t off = 0;
  auto AL = [&](size_t n) { float* p = Wsp + off; off += n; return p; };
  float* hbuf  = AL((size_t)NN * 320);   // [N,320] concat features
  float* big   = AL((size_t)NN * 256);   // xf pre-BN, then q|k|v|s (4x N*64), then agg
  float* small = AL((size_t)NN * 64);    // xl pre-BN, then attention out / gate result
  float* nbuf  = AL((size_t)NN * DE);    // [N,128]
  float* logit = AL(ESN);
  float* wexp  = AL(ESN);
  float* denom = AL(NN);
  float* nmaxf = AL(NN);                 // unsigned keys
  float* meanb = AL(1024);
  float* rstdb = AL(1024);
  float* score = AL(NN);
  float* kth   = AL(BGR);
  float* gbuf  = AL((size_t)BGR * DE);
  float* Xl    = AL((size_t)BGR * 1024);
  float* hg    = AL(128);
  float* hi    = AL(256);
  float* ha    = AL(256);
  float* xx    = AL((size_t)BGR * D2);
  float* xx2   = AL((size_t)BGR * D2);
  float* qkv   = AL((size_t)BGR * 3 * D2);
  float* scb   = AL((size_t)BGR * BGR);
  float* attn  = AL((size_t)BGR * D2);
  float* proj  = AL((size_t)BGR * D2);
  float* ffb   = AL((size_t)BGR * 2048);

  // ---- node feature + label branches: GEMM -> BN(batch stats) -> lrelu -> concat ----
  gemm(stream, x, DIN + DLBL, 0, PP(0), D2, 0, PP(1), big, D2, NN, D2, DIN, 0);
  colstats<<<D2, 256, 0, stream>>>(big, D2, NN, meanb, rstdb);
  norm_apply<<<ceil_div(NN * D2, 256), 256, 0, stream>>>(big, D2, NN, meanb, rstdb, PP(2), PP(3), hbuf, 320, 0);

  gemm(stream, x, DIN + DLBL, DIN, PP(4), DH, 0, PP(5), small, DH, NN, DH, DLBL, 0);
  colstats<<<DH, 256, 0, stream>>>(small, DH, NN, meanb, rstdb);
  norm_apply<<<ceil_div(NN * DH, 256), 256, 0, stream>>>(small, DH, NN, meanb, rstdb, PP(6), PP(7), hbuf, 320, D2);

  // ---- two TransformerConv branches (spatial / temporal) ----
  float* qb = big;
  float* kb = big + (size_t)NN * DH;
  float* vb = big + (size_t)2 * NN * DH;
  float* sb = big + (size_t)3 * NN * DH;
  for (int tag = 0; tag < 2; ++tag) {
    const int pb = tag == 0 ? 8 : 18;
    const int* src = tag == 0 ? ei : tadj;
    const int* dst = src + (tag == 0 ? ESN : ETN);
    const float* ew = tag == 0 ? (eemb + 3) : tw;
    const int ews = tag == 0 ? 4 : 1;
    const int E = tag == 0 ? ESN : ETN;

    gemm(stream, hbuf, 320, 0, PP(pb + 0), DH, 0, PP(pb + 1), qb, DH, NN, DH, 320, 0);
    gemm(stream, hbuf, 320, 0, PP(pb + 2), DH, 0, PP(pb + 3), kb, DH, NN, DH, 320, 0);
    gemm(stream, hbuf, 320, 0, PP(pb + 4), DH, 0, PP(pb + 5), vb, DH, NN, DH, 320, 0);
    gemm(stream, hbuf, 320, 0, PP(pb + 7), DH, 0, PP(pb + 8), sb, DH, NN, DH, 320, 0);

    fill(stream, nmaxf, 0.0f, NN);             // key 0 < key of any real logit
    fill(stream, denom, 0.0f, NN);
    fill(stream, small, 0.0f, NN * DH);        // attention output accumulator
    edge_logit<<<ceil_div(E, 8), 256, 0, stream>>>(src, dst, qb, kb, ew, ews, PP(pb + 6),
                                                   logit, (unsigned*)nmaxf, E);
    edge_expsum<<<ceil_div(E, 256), 256, 0, stream>>>(dst, logit, (const unsigned*)nmaxf,
                                                      wexp, denom, E);
    edge_scatter<<<ceil_div(E, 8), 256, 0, stream>>>(src, dst, vb, ew, ews, PP(pb + 6),
                                                     wexp, denom, small, E);
    beta_gate<<<ceil_div(NN, 8), 256, 0, stream>>>(small, sb, PP(pb + 9), small, NN);
    colstats<<<DH, 256, 0, stream>>>(small, DH, NN, meanb, rstdb);
    norm_apply<<<ceil_div(NN * DH, 256), 256, 0, stream>>>(small, DH, NN, meanb, rstdb,
                                                           nullptr, nullptr, nbuf, DE, tag * DH);
  }

  // ---- SAGPooling (GraphConv score, top-80 of 100 per graph, masked max readout) ----
  float* agg = big;  // q/k/v/s no longer needed
  fill(stream, agg, 0.0f, NN * DE);
  pool_agg<<<ceil_div(ESN, 8), 256, 0, stream>>>(ei, ei + ESN, nbuf, agg, ESN);
  pool_score<<<ceil_div(NN, 8), 256, 0, stream>>>(agg, nbuf, PP(28), PP(29), PP(30), score, NN);
  pool_kth<<<BGR, 128, 0, stream>>>(score, kth);
  pool_readout<<<BGR, 128, 0, stream>>>(nbuf, score, kth, gbuf);

  // ---- graph LSTM (128) ----
  gemm(stream, gbuf, DE, 0, PP(59), 4 * DE, 0, nullptr, Xl, 4 * DE, BGR, 4 * DE, DE, 0);
  lstm_seq<<<1, 512, 0, stream>>>(Xl, PP(60), PP(61), PP(62), BGR, DE, hg);

  // ---- two transformer encoders + LSTM(256) ----
  for (int tag = 0; tag < 2; ++tag) {
    const float* feat = tag == 0 ? imgf : attf;
    const int fcw = tag == 0 ? 31 : 33;
    const int eb  = tag == 0 ? 35 : 47;
    const int lb  = tag == 0 ? 63 : 67;
    float* hout   = tag == 0 ? hi : ha;

    gemm(stream, feat, DIN, 0, PP(fcw), D2, 0, PP(fcw + 1), xx, D2, BGR, D2, DIN, 0);
    gemm(stream, xx, D2, 0, PP(eb + 0), 3 * D2, 0, PP(eb + 1), qkv, 3 * D2, BGR, 3 * D2, D2, 0);
    // scores = q @ k^T (k slice accessed as [N,K] row-major => transB)
    gemm(stream, qkv, 3 * D2, 0, qkv + D2, 3 * D2, 1, nullptr, scb, BGR, BGR, BGR, D2, 0);
    softmax_rows<<<BGR, 256, 0, stream>>>(scb, BGR, 1.0f / 16.0f);
    gemm(stream, scb, BGR, 0, qkv + 2 * D2, 3 * D2, 0, nullptr, attn, D2, BGR, D2, BGR, 0);
    gemm(stream, attn, D2, 0, PP(eb + 2), D2, 0, PP(eb + 3), proj, D2, BGR, D2, D2, 0);
    resid_ln<<<BGR, D2, 0, stream>>>(xx, proj, PP(eb + 8), PP(eb + 9), xx2);
    gemm(stream, xx2, D2, 0, PP(eb + 4), 2048, 0, PP(eb + 5), ffb, 2048, BGR, 2048, D2, 1);
    gemm(stream, ffb, 2048, 0, PP(eb + 6), D2, 0, PP(eb + 7), proj, D2, BGR, D2, 2048, 0);
    resid_ln<<<BGR, D2, 0, stream>>>(xx2, proj, PP(eb + 10), PP(eb + 11), xx);

    gemm(stream, xx, D2, 0, PP(lb + 0), 4 * D2, 0, nullptr, Xl, 4 * D2, BGR, 4 * D2, D2, 0);
    lstm_seq<<<1, 1024, 0, stream>>>(Xl, PP(lb + 1), PP(lb + 2), PP(lb + 3), BGR, D2, hout);
  }

  // ---- fusion head ----
  final_fc<<<1, 128, 0, stream>>>(hg, hi, ha, PP(71), PP(72), PP(73), PP(74), (float*)d_out);
}
